// OLMoEDecoderLayer_82738249990936
// MI455X (gfx1250) — compile-verified, matrix operand order checked by reading, not run on hardware
//
#include <hip/hip_runtime.h>
#include <hip/hip_bf16.h>
#include <cstdint>
#include <cstddef>

// ---------------- problem constants (from reference) ----------------
constexpr int H_  = 1024;
constexpr int NH_ = 16;
constexpr int HD_ = 64;
constexpr int I_  = 1024;
constexpr int E_  = 16;
constexpr int S_  = 2048;   // sequence length
constexpr int T_  = 2048;   // tokens (B=1)

// ---------------- WMMA / TDM types (CDNA5 / gfx1250, wave32) ----------------
typedef __attribute__((ext_vector_type(16))) __bf16        v16bf;
typedef __attribute__((ext_vector_type(8)))  float         v8f;
typedef __attribute__((ext_vector_type(8)))  unsigned int  v8u;
typedef __attribute__((ext_vector_type(4)))  unsigned int  tdm_u32x4;
typedef __attribute__((ext_vector_type(8)))  int           tdm_i32x8;
typedef __attribute__((ext_vector_type(4)))  int           tdm_i32x4;

__device__ __forceinline__ unsigned short f2bf(float f) {
  unsigned int u = __float_as_uint(f);
  unsigned int r = u + 0x7FFFu + ((u >> 16) & 1u);   // round-to-nearest-even
  return (unsigned short)(r >> 16);
}

// ---------------- Tensor Data Mover: 2D strided tile -> LDS ----------------
// Stages a tile of `tileRows` rows x 64 bf16 from global (row stride `ld`
// elements) into LDS at `ldsAddr`. TDM pad feature inserts 4 DWORDs (16B)
// after every 32 DWORDs (64 bf16) -> LDS row stride 72 elements (bank-spread).
// Rows beyond `rows` (tensor_dim1) are zero-filled by TDM OOB handling.
__device__ __forceinline__ void tdm_load_tile64(unsigned ldsAddr,
                                                const unsigned short* gptr,
                                                int ld, int rows, int tileRows) {
  unsigned long long ga = (unsigned long long)(uintptr_t)gptr;
  tdm_u32x4 g0;
  g0[0] = 1u;                                              // count=1, user desc
  g0[1] = ldsAddr;                                         // LDS byte address
  g0[2] = (unsigned)(ga & 0xFFFFFFFFu);                    // global_addr[31:0]
  g0[3] = (unsigned)((ga >> 32) & 0x01FFFFFFu) | 0x80000000u; // [56:32] | type=2
  tdm_i32x8 g1;
  // data_size=2B (code 1), pad_enable, pad_interval=32DW (code 4), pad_amount=4DW (code 3)
  g1[0] = (int)(0x00010000u | (1u << 20) | (4u << 22) | (3u << 25));
  g1[1] = (int)(64u << 16);                                // tensor_dim0 = 64
  g1[2] = (int)(((unsigned)rows & 0xFFFFu) << 16);         // tensor_dim1 lo
  g1[3] = (int)((((unsigned)rows >> 16) & 0xFFFFu) | (64u << 16)); // dim1 hi | tile_dim0=64
  g1[4] = (int)((unsigned)tileRows & 0xFFFFu);             // tile_dim1, tile_dim2=0
  g1[5] = (int)(unsigned)ld;                               // tensor_dim0_stride lo32
  g1[6] = 0;                                               // stride hi, dim1_stride lo
  g1[7] = 0;
  tdm_i32x4 z4 = {0, 0, 0, 0};
#if defined(__clang_major__) && (__clang_major__ >= 23)
  tdm_i32x8 z8 = {0, 0, 0, 0, 0, 0, 0, 0};
  __builtin_amdgcn_tensor_load_to_lds(g0, g1, z4, z4, z8, 0);
#else
  __builtin_amdgcn_tensor_load_to_lds(g0, g1, z4, z4, 0);
#endif
}

// Load one 16x32 bf16 WMMA fragment from a padded LDS tile (row stride 72).
// Per ISA layout: lanes 0-15 hold row m=lane, K = [0..7],[16..23];
//                 lanes 16-31 hold row m=lane-16, K = [8..15],[24..31].
__device__ __forceinline__ v16bf load_frag_lds(const unsigned short* __restrict__ base,
                                               int lr0, int ks, int lane) {
  int m    = lane & 15;
  int half = lane >> 4;
  const unsigned short* p = base + (lr0 + m) * 72 + ks + half * 8;
  uint4 lo = *(const uint4*)p;            // ds_load_b128
  uint4 hi = *(const uint4*)(p + 16);
  v8u raw;
  raw[0] = lo.x; raw[1] = lo.y; raw[2] = lo.z; raw[3] = lo.w;
  raw[4] = hi.x; raw[5] = hi.y; raw[6] = hi.z; raw[7] = hi.w;
  return __builtin_bit_cast(v16bf, raw);
}

// ---------------- generic TN GEMM: C[M,N] = A[M,K] * B[N,K]^T ----------------
// TDM double-buffered LDS staging; block tile 128x64; 8 waves, wave tile 32x32.
// Requires Kd % 64 == 0 and M % 128 == 0 (true for all call sites here).
__global__ void __launch_bounds__(256)
gemm_tn_bf16_kernel(const unsigned short* __restrict__ A, int lda,
                    const unsigned short* __restrict__ B, int ldb,
                    float* __restrict__ C, unsigned short* __restrict__ Cb, int ldc,
                    int M, int N, int Kd) {
  __shared__ __align__(16) unsigned short ldsA[2][128 * 72];
  __shared__ __align__(16) unsigned short ldsB[2][64 * 72];

  int lane = threadIdx.x & 31;
  int wave = threadIdx.x >> 5;
  int wm = wave & 3, wn = wave >> 2;
  int rowBlk = blockIdx.x * 128;
  int colBlk = blockIdx.y * 64;
  int row0 = rowBlk + wm * 32;
  int col0 = colBlk + wn * 32;

  v8f acc[2][2];
  for (int a = 0; a < 2; ++a)
    for (int b = 0; b < 2; ++b)
      for (int i = 0; i < 8; ++i) acc[a][b][i] = 0.f;

  int rowsA = M - rowBlk; if (rowsA > 128) rowsA = 128;
  int rowsB = N - colBlk; if (rowsB > 64)  rowsB = 64;

  int nIter = Kd >> 6;
  if (wave == 0) {
    tdm_load_tile64((unsigned)(uintptr_t)&ldsA[0][0], A + (size_t)rowBlk * lda, lda, rowsA, 128);
    tdm_load_tile64((unsigned)(uintptr_t)&ldsB[0][0], B + (size_t)colBlk * ldb, ldb, rowsB, 64);
  }

  int buf = 0;
  for (int it = 0; it < nIter; ++it) {
    if (wave == 0) {
      if (it + 1 < nIter) {
        int k0 = (it + 1) << 6;
        tdm_load_tile64((unsigned)(uintptr_t)&ldsA[buf ^ 1][0],
                        A + (size_t)rowBlk * lda + k0, lda, rowsA, 128);
        tdm_load_tile64((unsigned)(uintptr_t)&ldsB[buf ^ 1][0],
                        B + (size_t)colBlk * ldb + k0, ldb, rowsB, 64);
        __builtin_amdgcn_s_wait_tensorcnt(2);   // retire stage(it) pair (in-order)
      } else {
        __builtin_amdgcn_s_wait_tensorcnt(0);
      }
    }
    __syncthreads();   // stage(it) visible to all waves

    const unsigned short* tA = &ldsA[buf][0];
    const unsigned short* tB = &ldsB[buf][0];
    for (int ks = 0; ks < 64; ks += 32) {
      v16bf a0 = load_frag_lds(tA, wm * 32,      ks, lane);
      v16bf a1 = load_frag_lds(tA, wm * 32 + 16, ks, lane);
      v16bf b0 = load_frag_lds(tB, wn * 32,      ks, lane);
      v16bf b1 = load_frag_lds(tB, wn * 32 + 16, ks, lane);
      acc[0][0] = __builtin_amdgcn_wmma_f32_16x16x32_bf16(false, a0, false, b0, (short)0, acc[0][0], false, false);
      acc[0][1] = __builtin_amdgcn_wmma_f32_16x16x32_bf16(false, a0, false, b1, (short)0, acc[0][1], false, false);
      acc[1][0] = __builtin_amdgcn_wmma_f32_16x16x32_bf16(false, a1, false, b0, (short)0, acc[1][0], false, false);
      acc[1][1] = __builtin_amdgcn_wmma_f32_16x16x32_bf16(false, a1, false, b1, (short)0, acc[1][1], false, false);
    }
    __syncthreads();   // buffer consumed; safe for TDM(it+2) to overwrite
    buf ^= 1;
  }

  // C/D layout: VGPR r -> (M = r + 8*half, N = lane&15) within each 16x16 tile
  int half = lane >> 4;
  for (int tm = 0; tm < 2; ++tm) {
    for (int tn = 0; tn < 2; ++tn) {
      int n = col0 + tn * 16 + (lane & 15);
      if (n >= N) continue;
      int rbase = row0 + tm * 16 + half * 8;
      for (int r = 0; r < 8; ++r) {
        int rr = rbase + r;
        if (rr >= M) break;
        float v = acc[tm][tn][r];
        if (Cb) Cb[(size_t)rr * ldc + n] = f2bf(v);
        else    C [(size_t)rr * ldc + n] = v;
      }
    }
  }
}

// ---------------- elementwise / reduction kernels ----------------
__global__ void convert_bf16_kernel(const float* __restrict__ in,
                                    unsigned short* __restrict__ out, size_t n) {
  size_t i = (size_t)blockIdx.x * blockDim.x + threadIdx.x;
  if (i < n) out[i] = f2bf(in[i]);
}

__global__ void add_kernel(const float* __restrict__ a, const float* __restrict__ b,
                           float* __restrict__ c, size_t n) {
  size_t i = (size_t)blockIdx.x * blockDim.x + threadIdx.x;
  if (i < n) c[i] = a[i] + b[i];
}

// RMSNorm over H=1024; one block per token; writes bf16 and/or f32.
__global__ void rmsnorm_kernel(const float* __restrict__ x, const float* __restrict__ w,
                               unsigned short* __restrict__ outb, float* __restrict__ outf) {
  int t = blockIdx.x;
  const float* xr = x + (size_t)t * H_;
  __shared__ float red[256];
  int tid = threadIdx.x;
  float s = 0.f;
  for (int i = tid; i < H_; i += 256) { float v = xr[i]; s += v * v; }
  red[tid] = s; __syncthreads();
  for (int st = 128; st > 0; st >>= 1) {
    if (tid < st) red[tid] += red[tid + st];
    __syncthreads();
  }
  float inv = rsqrtf(red[0] / (float)H_ + 1e-6f);
  for (int i = tid; i < H_; i += 256) {
    float v = xr[i] * inv * w[i];
    if (outf) outf[(size_t)t * H_ + i] = v;
    if (outb) outb[(size_t)t * H_ + i] = f2bf(v);
  }
}

// Row softmax of scores (scaled by HD^-0.5 = 0.125) -> bf16 probabilities.
__global__ void softmax_row_kernel(const float* __restrict__ sc,
                                   unsigned short* __restrict__ P) {
  int r = blockIdx.x;
  const float* row = sc + (size_t)r * S_;
  __shared__ float red[256];
  int tid = threadIdx.x;
  const float scale = 0.125f;
  float mx = -3.0e38f;
  for (int i = tid; i < S_; i += 256) mx = fmaxf(mx, row[i] * scale);
  red[tid] = mx; __syncthreads();
  for (int st = 128; st > 0; st >>= 1) {
    if (tid < st) red[tid] = fmaxf(red[tid], red[tid + st]);
    __syncthreads();
  }
  mx = red[0]; __syncthreads();
  float sum = 0.f;
  for (int i = tid; i < S_; i += 256) sum += expf(row[i] * scale - mx);
  red[tid] = sum; __syncthreads();
  for (int st = 128; st > 0; st >>= 1) {
    if (tid < st) red[tid] += red[tid + st];
    __syncthreads();
  }
  float inv = 1.f / red[0];
  for (int i = tid; i < S_; i += 256)
    P[(size_t)r * S_ + i] = f2bf(expf(row[i] * scale - mx) * inv);
}

// V head transpose: Vt[hd, t] = V[t, head*64 + hd]  (bf16, K-major for GEMM B)
__global__ void transpose_v_kernel(const unsigned short* __restrict__ vb,
                                   unsigned short* __restrict__ Vt, int head) {
  int idx = blockIdx.x * 256 + threadIdx.x;
  if (idx >= HD_ * S_) return;
  int t = idx >> 6, hd = idx & 63;
  Vt[(size_t)hd * S_ + t] = vb[(size_t)t * H_ + head * HD_ + hd];
}

// Top-4 routing + dense combine weights + aux-loss statistics.
__global__ void routing_kernel(const float* __restrict__ logits,
                               float* __restrict__ wdense,
                               float* __restrict__ psum,
                               float* __restrict__ counts) {
  int t = blockIdx.x * blockDim.x + threadIdx.x;
  if (t >= T_) return;
  float l[E_];
  for (int e = 0; e < E_; ++e) l[e] = logits[(size_t)t * E_ + e];

  // full softmax (for aux loss)
  float mx = -3.0e38f;
  for (int e = 0; e < E_; ++e) mx = fmaxf(mx, l[e]);
  float p[E_], se = 0.f;
  for (int e = 0; e < E_; ++e) { p[e] = expf(l[e] - mx); se += p[e]; }
  for (int e = 0; e < E_; ++e) atomicAdd(&psum[e], p[e] / se);

  // top-4 (first-max tie-break, matches lax.top_k)
  float lv[E_];
  for (int e = 0; e < E_; ++e) lv[e] = l[e];
  int   sel[4]; float tv[4];
  for (int k = 0; k < 4; ++k) {
    int bj = 0; float bv = lv[0];
    for (int e = 1; e < E_; ++e) if (lv[e] > bv) { bv = lv[e]; bj = e; }
    sel[k] = bj; tv[k] = bv; lv[bj] = -3.0e38f;
  }
  float m2 = tv[0], s2 = 0.f, rw[4];
  for (int k = 0; k < 4; ++k) { rw[k] = expf(tv[k] - m2); s2 += rw[k]; }
  for (int e = 0; e < E_; ++e) wdense[(size_t)t * E_ + e] = 0.f;
  for (int k = 0; k < 4; ++k) {
    wdense[(size_t)t * E_ + sel[k]] += rw[k] / s2;
    atomicAdd(&counts[sel[k]], 1.0f);
  }
}

// h = silu(g) * u  -> bf16
__global__ void silu_mul_kernel(const float* __restrict__ g, const float* __restrict__ u,
                                unsigned short* __restrict__ hb, size_t n) {
  size_t i = (size_t)blockIdx.x * blockDim.x + threadIdx.x;
  if (i >= n) return;
  float gg = g[i];
  hb[i] = f2bf((gg / (1.f + expf(-gg))) * u[i]);
}

// acc += wdense[t, e] * d
__global__ void moe_accum_kernel(float* __restrict__ acc, const float* __restrict__ d,
                                 const float* __restrict__ wdense, int e) {
  size_t i = (size_t)blockIdx.x * blockDim.x + threadIdx.x;
  if (i >= (size_t)T_ * H_) return;
  size_t t = i >> 10;   // H = 1024
  acc[i] += wdense[t * E_ + e] * d[i];
}

__global__ void aux_kernel(const float* __restrict__ psum, const float* __restrict__ counts,
                           float* __restrict__ out) {
  if (blockIdx.x == 0 && threadIdx.x == 0) {
    float a = 0.f;
    for (int e = 0; e < E_; ++e) a += (psum[e] / (float)T_) * (counts[e] / (float)T_);
    out[(size_t)T_ * H_] = a;
  }
}

// ---------------- host orchestration ----------------
extern "C" void kernel_launch(void* const* d_in, const int* in_sizes, int n_in,
                              void* d_out, int out_size, void* d_ws, size_t ws_size,
                              hipStream_t stream) {
  (void)in_sizes; (void)n_in; (void)out_size; (void)ws_size;
  const float* x      = (const float*)d_in[0];
  const float* wq     = (const float*)d_in[1];
  const float* wk     = (const float*)d_in[2];
  const float* wv     = (const float*)d_in[3];
  const float* wo     = (const float*)d_in[4];
  const float* gate_w = (const float*)d_in[5];
  const float* w_gate = (const float*)d_in[6];
  const float* w_up   = (const float*)d_in[7];
  const float* w_down = (const float*)d_in[8];
  const float* ln_in  = (const float*)d_in[9];
  const float* ln_at  = (const float*)d_in[10];
  const float* ln_ff  = (const float*)d_in[11];
  float* out = (float*)d_out;

  // ---- carve workspace ----
  char* base = (char*)d_ws;
  size_t off = 0;
  auto alloc = [&](size_t bytes) -> void* {
    off = (off + 255) & ~(size_t)255;
    void* p = base + off;
    off += bytes;
    return p;
  };
  typedef unsigned short us;
  us* wqb  = (us*)alloc((size_t)H_ * H_ * 2);
  us* wkb  = (us*)alloc((size_t)H_ * H_ * 2);
  us* wvb  = (us*)alloc((size_t)H_ * H_ * 2);
  us* wob  = (us*)alloc((size_t)H_ * H_ * 2);
  us* gtb  = (us*)alloc((size_t)E_ * H_ * 2);
  us* wgb  = (us*)alloc((size_t)E_ * I_ * H_ * 2);
  us* wub  = (us*)alloc((size_t)E_ * I_ * H_ * 2);
  us* wdb  = (us*)alloc((size_t)E_ * H_ * I_ * 2);
  us* xn   = (us*)alloc((size_t)T_ * H_ * 2);
  us* qb   = (us*)alloc((size_t)T_ * H_ * 2);
  us* kb   = (us*)alloc((size_t)T_ * H_ * 2);
  us* vb   = (us*)alloc((size_t)T_ * H_ * 2);
  float* scores = (float*)alloc((size_t)S_ * S_ * 4);
  us* Pb   = (us*)alloc((size_t)S_ * S_ * 2);
  us* Vt   = (us*)alloc((size_t)HD_ * S_ * 2);
  float* attnf = (float*)alloc((size_t)T_ * H_ * 4);
  us* attnb    = (us*)alloc((size_t)T_ * H_ * 2);
  float* attno = (float*)alloc((size_t)T_ * H_ * 4);
  float* hres  = (float*)alloc((size_t)T_ * H_ * 4);
  us* xn2  = (us*)alloc((size_t)T_ * H_ * 2);
  float* logits = (float*)alloc((size_t)T_ * E_ * 4);
  float* wdense = (float*)alloc((size_t)T_ * E_ * 4);
  float* psum   = (float*)alloc((size_t)E_ * 4);
  float* counts = (float*)alloc((size_t)E_ * 4);
  float* gbuf = (float*)alloc((size_t)T_ * I_ * 4);
  float* ubuf = (float*)alloc((size_t)T_ * I_ * 4);
  us*    hb   = (us*)alloc((size_t)T_ * I_ * 2);
  float* dbuf = (float*)alloc((size_t)T_ * H_ * 4);
  float* macc = (float*)alloc((size_t)T_ * H_ * 4);
  float* h2   = (float*)alloc((size_t)T_ * H_ * 4);

  hipMemsetAsync(psum,   0, (size_t)E_ * 4, stream);
  hipMemsetAsync(counts, 0, (size_t)E_ * 4, stream);
  hipMemsetAsync(macc,   0, (size_t)T_ * H_ * 4, stream);

  auto cvt = [&](const float* src, us* dst, size_t n) {
    int blocks = (int)((n + 255) / 256);
    convert_bf16_kernel<<<blocks, 256, 0, stream>>>(src, dst, n);
  };
  auto gemm = [&](const us* A, int lda, const us* B, int ldb,
                  float* C, us* Cb, int ldc, int M, int N, int Kd) {
    dim3 g((M + 127) / 128, (N + 63) / 64);
    gemm_tn_bf16_kernel<<<g, 256, 0, stream>>>(A, lda, B, ldb, C, Cb, ldc, M, N, Kd);
  };

  // ---- weight conversion (f32 -> bf16) ----
  cvt(wq, wqb, (size_t)H_ * H_);
  cvt(wk, wkb, (size_t)H_ * H_);
  cvt(wv, wvb, (size_t)H_ * H_);
  cvt(wo, wob, (size_t)H_ * H_);
  cvt(gate_w, gtb, (size_t)E_ * H_);
  cvt(w_gate, wgb, (size_t)E_ * I_ * H_);
  cvt(w_up,   wub, (size_t)E_ * I_ * H_);
  cvt(w_down, wdb, (size_t)E_ * H_ * I_);

  // ---- attention ----
  rmsnorm_kernel<<<T_, 256, 0, stream>>>(x, ln_in, xn, nullptr);
  gemm(xn, H_, wqb, H_, nullptr, qb, H_, T_, H_, H_);   // Q (bf16 out)
  gemm(xn, H_, wkb, H_, nullptr, kb, H_, T_, H_, H_);   // K
  gemm(xn, H_, wvb, H_, nullptr, vb, H_, T_, H_, H_);   // V

  for (int h = 0; h < NH_; ++h) {
    // scores = Q_h * K_h^T  (M=N=2048, K=64)
    gemm(qb + h * HD_, H_, kb + h * HD_, H_, scores, nullptr, S_, S_, S_, HD_);
    softmax_row_kernel<<<S_, 256, 0, stream>>>(scores, Pb);
    transpose_v_kernel<<<(HD_ * S_ + 255) / 256, 256, 0, stream>>>(vb, Vt, h);
    // O_h = P * V_h   (M=2048, N=64, K=2048)
    gemm(Pb, S_, Vt, S_, attnf + h * HD_, nullptr, H_, T_, HD_, S_);
  }

  cvt(attnf, attnb, (size_t)T_ * H_);
  gemm(attnb, H_, wob, H_, attno, nullptr, H_, T_, H_, H_);   // O projection
  add_kernel<<<(T_ * H_ + 255) / 256, 256, 0, stream>>>(x, attno, hres, (size_t)T_ * H_);

  // ---- MoE ----
  rmsnorm_kernel<<<T_, 256, 0, stream>>>(hres, ln_at, xn2, nullptr);
  gemm(xn2, H_, gtb, H_, logits, nullptr, E_, T_, E_, H_);    // router logits
  routing_kernel<<<(T_ + 255) / 256, 256, 0, stream>>>(logits, wdense, psum, counts);

  for (int e = 0; e < E_; ++e) {
    const us* gw = wgb + (size_t)e * I_ * H_;
    const us* uw = wub + (size_t)e * I_ * H_;
    const us* dw = wdb + (size_t)e * H_ * I_;
    gemm(xn2, H_, gw, H_, gbuf, nullptr, I_, T_, I_, H_);
    gemm(xn2, H_, uw, H_, ubuf, nullptr, I_, T_, I_, H_);
    silu_mul_kernel<<<(T_ * I_ + 255) / 256, 256, 0, stream>>>(gbuf, ubuf, hb, (size_t)T_ * I_);
    gemm(hb, I_, dw, I_, dbuf, nullptr, H_, T_, H_, I_);
    moe_accum_kernel<<<(T_ * H_ + 255) / 256, 256, 0, stream>>>(macc, dbuf, wdense, e);
  }

  // ---- final residual + norm + aux ----
  add_kernel<<<(T_ * H_ + 255) / 256, 256, 0, stream>>>(hres, macc, h2, (size_t)T_ * H_);
  rmsnorm_kernel<<<T_, 256, 0, stream>>>(h2, ln_ff, nullptr, out);
  aux_kernel<<<1, 64, 0, stream>>>(psum, counts, out);
}
